// TemporalGATLayer_85942295593119
// MI455X (gfx1250) — matrix operand order repaired and checked by reading, not compiled, not run on hardware
//
#include <hip/hip_runtime.h>

// ---------------------------------------------------------------------------
// TemporalGATLayer for MI455X (gfx1250, wave32, WMMA + async-to-LDS copies).
// GEMMs: v_wmma_f32_16x16x32_f16 (f16 A/B, fp32 accum).
// Weights are pre-converted to f16 once (k_prep) and then staged into LDS with
// global_load_async_to_lds_b128 / s_wait_asynccnt (CDNA5 async path), issued
// early and waited late so the copies overlap the fp32 alpha/softmax + tile
// construction VALU work. One-hot adjacency inputs are never read; the
// scatter-add is a per-target-node block reduction (no atomics).
// ---------------------------------------------------------------------------

typedef __attribute__((ext_vector_type(16))) _Float16 v16h;
typedef __attribute__((ext_vector_type(8)))  float    v8f;

#define BB 32
#define NN 64
#define FF 128
#define EE 4032
#define HH 4
#define DD 64

__device__ __forceinline__ v8f wmma16(v16h a, v16h b, v8f c) {
  // D = A(16x32 f16) * B(32x16 f16) + C(16x16 f32)
  return __builtin_amdgcn_wmma_f32_16x16x32_f16(
      /*neg_a=*/false, a, /*neg_b=*/false, b,
      /*c_mod=*/(short)0, c, /*reuse_a=*/false, /*reuse_b=*/false);
}

// Async copy 16B/lane from global (SGPR base + per-lane byte offset, GVS mode)
// into LDS (per-lane LDS byte address in VGPR). Tracked by ASYNCcnt.
__device__ __forceinline__ void async_b128(unsigned lds_off, const void* gbase,
                                           int goff) {
  asm volatile("global_load_async_to_lds_b128 %0, %1, %2"
               :: "v"(lds_off), "v"(goff), "s"(gbase)
               : "memory");
}
__device__ __forceinline__ void wait_async0() {
  asm volatile("s_wait_asynccnt 0x0" ::: "memory");
}
// Generic pointer to LDS: shared-aperture flat address keeps the LDS byte
// offset in addr[31:0] (ISA aperture table), which is what VDST needs.
__device__ __forceinline__ unsigned lds_off_of(const void* p) {
  return (unsigned)(size_t)p;
}

// A fragment: 16x32 f16 tile, row-major with leading dim `ld`.
// ISA layout: lane<16 row=lane, K in {8h..8h+7} U {16+8h..16+8h+7}, h=lane>>4.
__device__ __forceinline__ v16h frag_a(const _Float16* t, int ld, int lane) {
  const int r = lane & 15, hh = lane >> 4;
  const _Float16* p = t + r * ld + 8 * hh;
  v16h f;
#pragma unroll
  for (int e = 0; e < 8; ++e) f[e] = p[e];
#pragma unroll
  for (int e = 0; e < 8; ++e) f[8 + e] = p[16 + e];
  return f;
}

// B fragment for D = X @ W^T: W stored row-major [out][k] == B^T, so column
// c = out index (lane&15), K in {8h..8h+7} U {16+8h..16+8h+7}, h=lane>>4.
__device__ __forceinline__ v16h frag_b(const _Float16* t, int ld, int lane) {
  const int c = lane & 15, hh = lane >> 4;
  const _Float16* p = t + c * ld + 8 * hh;
  v16h f;
#pragma unroll
  for (int e = 0; e < 8; ++e) f[e] = p[e];
#pragma unroll
  for (int e = 0; e < 8; ++e) f[8 + e] = p[16 + e];
  return f;
}

// ---------------------------------------------------------------------------
// K0: one-shot fp32 -> f16 conversion of all MLP weights into workspace.
// Layout (f16 elements): We1[8192] | We2[4096] | Wv1[8192] | Wv2[16384] |
// Wv3[8192]  -> 45056 elements total.
// ---------------------------------------------------------------------------
__global__ __launch_bounds__(256) void k_prep(const float* __restrict__ We1,
                                              const float* __restrict__ We2,
                                              const float* __restrict__ Wv1,
                                              const float* __restrict__ Wv2,
                                              const float* __restrict__ Wv3,
                                              _Float16* __restrict__ dst) {
  const int i = blockIdx.x * 256 + threadIdx.x;
  if (i < 8192)       dst[i] = (_Float16)We1[i];
  else if (i < 12288) dst[i] = (_Float16)We2[i - 8192];
  else if (i < 20480) dst[i] = (_Float16)Wv1[i - 12288];
  else if (i < 36864) dst[i] = (_Float16)Wv2[i - 20480];
  else if (i < 45056) dst[i] = (_Float16)Wv3[i - 36864];
}

// ---------------------------------------------------------------------------
// K1: v_proj = v_self @ W_proj^T.  [2048,128] x [128,256] -> P (f16).
// ---------------------------------------------------------------------------
__global__ __launch_bounds__(256) void k_proj(const float* __restrict__ v_self,
                                              const float* __restrict__ W_proj,
                                              _Float16* __restrict__ P) {
  __shared__ _Float16 As[64 * 128];
  __shared__ _Float16 Ws[64 * 128];
  const int tid = threadIdx.x;
  const int bm = blockIdx.x >> 2;  // 0..31
  const int bn = blockIdx.x & 3;   // 0..3
  const int row0 = bm * 64;
  const int col0 = bn * 64;
  for (int idx = tid; idx < 64 * 128; idx += 256) {
    As[idx] = (_Float16)v_self[(row0 + (idx >> 7)) * FF + (idx & 127)];
    Ws[idx] = (_Float16)W_proj[(col0 + (idx >> 7)) * FF + (idx & 127)];
  }
  __syncthreads();
  const int w = tid >> 5, lane = tid & 31;
  const int rt = w >> 1, ctb = (w & 1) * 2;
  v8f acc0 = {}, acc1 = {};
#pragma unroll
  for (int k0 = 0; k0 < 128; k0 += 32) {
    v16h a  = frag_a(As + (rt * 16) * 128 + k0, 128, lane);
    v16h b0 = frag_b(Ws + (ctb * 16) * 128 + k0, 128, lane);
    v16h b1 = frag_b(Ws + ((ctb + 1) * 16) * 128 + k0, 128, lane);
    acc0 = wmma16(a, b0, acc0);
    acc1 = wmma16(a, b1, acc1);
  }
  const int cin = lane & 15, rbase = 8 * (lane >> 4);
#pragma unroll
  for (int v = 0; v < 8; ++v) {
    const int r = row0 + rt * 16 + rbase + v;
    P[r * 256 + col0 + ctb * 16 + cin]       = (_Float16)acc0[v];
    P[r * 256 + col0 + (ctb + 1) * 16 + cin] = (_Float16)acc1[v];
  }
}

// ---------------------------------------------------------------------------
// K2: per-node attention logits: s = leaky(P[b,n,h,:] . a)/500  (fp32).
// ---------------------------------------------------------------------------
__global__ __launch_bounds__(256) void k_attn(const _Float16* __restrict__ P,
                                              const float* __restrict__ a_fwd,
                                              const float* __restrict__ a_bwd,
                                              float* __restrict__ s_fwd,
                                              float* __restrict__ s_bwd) {
  const int g = blockIdx.x * 256 + threadIdx.x;  // ((b*64+n)*4+h), 8192 total
  const int h = g & 3;
  const int rownode = g >> 2;
  const _Float16* p = P + rownode * 256 + h * 64;
  const float* af = a_fwd + h * 64;
  const float* ab = a_bwd + h * 64;
  float sf = 0.f, sb = 0.f;
#pragma unroll 8
  for (int d = 0; d < 64; ++d) {
    const float x = (float)p[d];
    sf += x * af[d];
    sb += x * ab[d];
  }
  sf = (sf >= 0.f ? sf : 0.2f * sf) * (1.0f / 500.0f);
  sb = (sb >= 0.f ? sb : 0.2f * sb) * (1.0f / 500.0f);
  s_fwd[g] = sf;
  s_bwd[g] = sb;
}

// ---------------------------------------------------------------------------
// K3 (dominant): one block per (b, target j, head h).
// Rows = 64 candidate sources i (i==j padded, alpha=0 -> contributes 0).
//   X[64x128] = [a_ij*h_src | a_ji*h_tgt]  -> WMMA We1 + relu -> T[64x64]
//   -> WMMA We2 + bias -> * alpha_ij -> reduce over 64 edge rows
//   -> v_social[b,j,h,:]. Weights async-copied to LDS, waited after X-build.
// ---------------------------------------------------------------------------
__global__ __launch_bounds__(256) void k_edge(const _Float16* __restrict__ P,
                                              const float* __restrict__ s_fwd,
                                              const float* __restrict__ s_bwd,
                                              const _Float16* __restrict__ We1h,
                                              const float* __restrict__ be1,
                                              const _Float16* __restrict__ We2h,
                                              const float* __restrict__ be2,
                                              float* __restrict__ v_social,
                                              float* __restrict__ alpha_out) {
  __shared__ _Float16 Xs[64 * 128];   // edge-input tile; reused as fp32 ew
  __shared__ _Float16 W1s[64 * 128];
  __shared__ _Float16 Ts[64 * 64];
  __shared__ _Float16 W2s[64 * 64];
  __shared__ float aI[64], aJ[64], b1s[64], b2s[64];
  __shared__ float red[4 * 64];
  float* ew = reinterpret_cast<float*>(Xs);  // 64x64 fp32, fits in Xs' 16KB

  const int tid = threadIdx.x;
  const int h = blockIdx.x & 3;
  const int j = (blockIdx.x >> 2) & 63;
  const int b = blockIdx.x >> 8;

  // --- issue async weight copies now; wait only after alpha + X build ---
  {
    const unsigned l1 = lds_off_of(&W1s[0]);
#pragma unroll
    for (int it = 0; it < 4; ++it) {       // 16KB of We1 (f16)
      const int off = (it * 256 + tid) * 16;
      async_b128(l1 + off, We1h, off);
    }
    const unsigned l2 = lds_off_of(&W2s[0]);
#pragma unroll
    for (int it = 0; it < 2; ++it) {       // 8KB of We2 (f16)
      const int off = (it * 256 + tid) * 16;
      async_b128(l2 + off, We2h, off);
    }
  }

  // --- stage 0: fp32 alpha (exp(exp()) softmax pair), write alpha output ---
  if (tid < 64) {
    const int i = tid;
    float ai = 0.f, aj = 0.f;
    if (i != j) {
      const float x = s_fwd[(b * 64 + i) * 4 + h];
      const float y = s_bwd[(b * 64 + j) * 4 + h];
      const float m = fmaxf(x, y);
      const float eij = __expf(__expf(x - m));
      const float eji = __expf(__expf(y - m));
      const float inv = 1.0f / (eij + eji);
      ai = eij * inv;
      aj = eji * inv;
      const int e = i * 63 + (j < i ? j : j - 1);
      alpha_out[(b * EE + e) * HH + h] = ai;
    }
    aI[i] = ai;
    aJ[i] = aj;
    b1s[i] = be1[i];
    b2s[i] = be2[i];
  }
  __syncthreads();  // aI/aJ visible for X build

  // --- stage 1: build X = [alpha_ij * h_src | alpha_ji * h_tgt] (f16) ---
  const _Float16* Pj = P + (b * 64 + j) * 256 + h * 64;
  for (int idx = tid; idx < 64 * 128; idx += 256) {
    const int r = idx >> 7, c = idx & 127;
    float v;
    if (c < 64) v = aI[r] * (float)P[(b * 64 + r) * 256 + h * 64 + c];
    else        v = aJ[r] * (float)Pj[c - 64];
    Xs[idx] = (_Float16)v;
  }
  wait_async0();    // weights landed in LDS
  __syncthreads();

  const int w = tid >> 5, lane = tid & 31;
  const int rt = w >> 1, ctb = (w & 1) * 2;
  const int cin = lane & 15, rbase = 8 * (lane >> 4);

  // --- GEMM1: X[64x128] @ We1^T -> relu -> T[64x64] f16 ---
  {
    v8f acc0 = {}, acc1 = {};
#pragma unroll
    for (int k0 = 0; k0 < 128; k0 += 32) {
      v16h a  = frag_a(Xs + (rt * 16) * 128 + k0, 128, lane);
      v16h b0 = frag_b(W1s + (ctb * 16) * 128 + k0, 128, lane);
      v16h b1 = frag_b(W1s + ((ctb + 1) * 16) * 128 + k0, 128, lane);
      acc0 = wmma16(a, b0, acc0);
      acc1 = wmma16(a, b1, acc1);
    }
    const int c0 = ctb * 16 + cin, c1 = c0 + 16;
    const float bb0 = b1s[c0], bb1 = b1s[c1];
#pragma unroll
    for (int v = 0; v < 8; ++v) {
      const int r = rt * 16 + rbase + v;
      Ts[r * 64 + c0] = (_Float16)fmaxf(acc0[v] + bb0, 0.f);
      Ts[r * 64 + c1] = (_Float16)fmaxf(acc1[v] + bb1, 0.f);
    }
  }
  __syncthreads();  // all Xs reads done; Xs region now reusable as ew

  // --- GEMM2: T @ We2^T + be2, scale rows by alpha_ij -> ew (fp32 LDS) ---
  {
    v8f d0 = {}, d1 = {};
#pragma unroll
    for (int k0 = 0; k0 < 64; k0 += 32) {
      v16h a  = frag_a(Ts + (rt * 16) * 64 + k0, 64, lane);
      v16h b0 = frag_b(W2s + (ctb * 16) * 64 + k0, 64, lane);
      v16h b1 = frag_b(W2s + ((ctb + 1) * 16) * 64 + k0, 64, lane);
      d0 = wmma16(a, b0, d0);
      d1 = wmma16(a, b1, d1);
    }
    const int c0 = ctb * 16 + cin, c1 = c0 + 16;
    const float bb0 = b2s[c0], bb1 = b2s[c1];
#pragma unroll
    for (int v = 0; v < 8; ++v) {
      const int r = rt * 16 + rbase + v;
      const float al = aI[r];
      ew[r * 64 + c0] = (d0[v] + bb0) * al;
      ew[r * 64 + c1] = (d1[v] + bb1) * al;
    }
  }
  __syncthreads();

  // --- scatter-as-reduction: v_social[b,j,h,d] = sum over 64 edge rows ---
  {
    const int d = tid & 63, q = tid >> 6;
    float s = 0.f;
#pragma unroll
    for (int r = 0; r < 16; ++r) s += ew[(q * 16 + r) * 64 + d];
    red[q * 64 + d] = s;
  }
  __syncthreads();
  if (tid < 64) {
    const float s = red[tid] + red[64 + tid] + red[128 + tid] + red[192 + tid];
    v_social[((b * 64 + j) * 4 + h) * 64 + tid] = s;
  }
}

// ---------------------------------------------------------------------------
// K4: node MLP chain per (b,n,h) row: 64 ->relu 128 ->relu 128 -> 64.
// Block: 64 rows. Weights async-staged into a reused 16KB LDS buffer
// (Wv2 done in two 64-output-column passes). LDS total = 56KB.
// ---------------------------------------------------------------------------
__global__ __launch_bounds__(256) void k_node(const float* __restrict__ vs,
                                              const _Float16* __restrict__ Wv1h,
                                              const float* __restrict__ bv1,
                                              const _Float16* __restrict__ Wv2h,
                                              const float* __restrict__ bv2,
                                              const _Float16* __restrict__ Wv3h,
                                              const float* __restrict__ bv3,
                                              float* __restrict__ out) {
  __shared__ _Float16 Xs[64 * 64];     //  8KB
  __shared__ _Float16 X1[64 * 128];    // 16KB
  __shared__ _Float16 X2[64 * 128];    // 16KB
  __shared__ _Float16 Wbuf[128 * 64];  // 16KB weight staging, reused per stage
  const int tid = threadIdx.x;
  const int row0 = blockIdx.x * 64;  // rows are ((b*64+n)*4+h)
  const unsigned lw = lds_off_of(&Wbuf[0]);

  // async stage Wv1 (128x64 f16 = 16KB) while converting the input tile
#pragma unroll
  for (int it = 0; it < 4; ++it) {
    const int off = (it * 256 + tid) * 16;
    async_b128(lw + off, Wv1h, off);
  }
  for (int idx = tid; idx < 64 * 64; idx += 256)
    Xs[idx] = (_Float16)vs[row0 * 64 + idx];
  wait_async0();
  __syncthreads();

  const int w = tid >> 5, lane = tid & 31;
  const int cin = lane & 15, rbase = 8 * (lane >> 4);

  // stage 1: 64 -> 128, relu   (Wbuf = Wv1 [128][64])
  {
    const int rt = w >> 1, ctb = (w & 1) * 4;
    v8f acc[4] = {};
#pragma unroll
    for (int k0 = 0; k0 < 64; k0 += 32) {
      v16h a = frag_a(Xs + (rt * 16) * 64 + k0, 64, lane);
#pragma unroll
      for (int t = 0; t < 4; ++t) {
        v16h bf = frag_b(Wbuf + ((ctb + t) * 16) * 64 + k0, 64, lane);
        acc[t] = wmma16(a, bf, acc[t]);
      }
    }
#pragma unroll
    for (int t = 0; t < 4; ++t) {
      const int col = (ctb + t) * 16 + cin;
      const float bias = bv1[col];
#pragma unroll
      for (int v = 0; v < 8; ++v) {
        const int r = rt * 16 + rbase + v;
        X1[r * 128 + col] = (_Float16)fmaxf(acc[t][v] + bias, 0.f);
      }
    }
  }
  __syncthreads();  // X1 visible; all stage-1 Wbuf reads complete

  // stage 2: 128 -> 128, relu; two passes of 64 output cols, each staging
  // one 64x128 half of Wv2 (16KB) into Wbuf.
#pragma unroll 1
  for (int p = 0; p < 2; ++p) {
    const _Float16* g = Wv2h + p * 64 * 128;
#pragma unroll
    for (int it = 0; it < 4; ++it) {
      const int off = (it * 256 + tid) * 16;
      async_b128(lw + off, g, off);
    }
    wait_async0();
    __syncthreads();  // half of Wv2 resident

    const int rt = w >> 1, cts = (w & 1) * 2;
    v8f a0 = {}, a1 = {};
#pragma unroll
    for (int k0 = 0; k0 < 128; k0 += 32) {
      v16h a  = frag_a(X1 + (rt * 16) * 128 + k0, 128, lane);
      v16h b0 = frag_b(Wbuf + (cts * 16) * 128 + k0, 128, lane);
      v16h b1 = frag_b(Wbuf + ((cts + 1) * 16) * 128 + k0, 128, lane);
      a0 = wmma16(a, b0, a0);
      a1 = wmma16(a, b1, a1);
    }
    const int c0 = p * 64 + cts * 16 + cin, c1 = c0 + 16;
    const float bb0 = bv2[c0], bb1 = bv2[c1];
#pragma unroll
    for (int v = 0; v < 8; ++v) {
      const int r = rt * 16 + rbase + v;
      X2[r * 128 + c0] = (_Float16)fmaxf(a0[v] + bb0, 0.f);
      X2[r * 128 + c1] = (_Float16)fmaxf(a1[v] + bb1, 0.f);
    }
    __syncthreads();  // pass reads of Wbuf done before next overwrite
  }

  // stage 3: 128 -> 64, write v_out (fp32)   (Wbuf = Wv3 [64][128])
#pragma unroll
  for (int it = 0; it < 4; ++it) {
    const int off = (it * 256 + tid) * 16;
    async_b128(lw + off, Wv3h, off);
  }
  wait_async0();
  __syncthreads();
  {
    const int rt = w >> 1, ctb = (w & 1) * 2;
    v8f acc0 = {}, acc1 = {};
#pragma unroll
    for (int k0 = 0; k0 < 128; k0 += 32) {
      v16h a  = frag_a(X2 + (rt * 16) * 128 + k0, 128, lane);
      v16h b0 = frag_b(Wbuf + (ctb * 16) * 128 + k0, 128, lane);
      v16h b1 = frag_b(Wbuf + ((ctb + 1) * 16) * 128 + k0, 128, lane);
      acc0 = wmma16(a, b0, acc0);
      acc1 = wmma16(a, b1, acc1);
    }
    const int c0 = ctb * 16 + cin, c1 = c0 + 16;
    const float bb0 = bv3[c0], bb1 = bv3[c1];
#pragma unroll
    for (int v = 0; v < 8; ++v) {
      const int r = row0 + rt * 16 + rbase + v;
      out[r * 64 + c0] = acc0[v] + bb0;  // == v_out[b, n, h*64 + c]
      out[r * 64 + c1] = acc1[v] + bb1;
    }
  }
}

// ---------------------------------------------------------------------------
extern "C" void kernel_launch(void* const* d_in, const int* in_sizes, int n_in,
                              void* d_out, int out_size, void* d_ws,
                              size_t ws_size, hipStream_t stream) {
  const float* v_self = (const float*)d_in[0];
  // d_in[1] = rel_rec, d_in[2] = rel_send: one-hot, intentionally unused.
  const float* W_proj = (const float*)d_in[3];
  const float* a_fwd  = (const float*)d_in[4];
  const float* a_bwd  = (const float*)d_in[5];
  const float* We1 = (const float*)d_in[6];
  const float* be1 = (const float*)d_in[7];
  const float* We2 = (const float*)d_in[8];
  const float* be2 = (const float*)d_in[9];
  const float* Wv1 = (const float*)d_in[10];
  const float* bv1 = (const float*)d_in[11];
  const float* Wv2 = (const float*)d_in[12];
  const float* bv2 = (const float*)d_in[13];
  const float* Wv3 = (const float*)d_in[14];
  const float* bv3 = (const float*)d_in[15];

  float* out_f     = (float*)d_out;
  float* v_out     = out_f;                      // [32,64,256]
  float* alpha_out = out_f + BB * NN * HH * DD;  // [32,4032,4]

  // workspace layout (fully rewritten every call; no cross-call state):
  char* ws = (char*)d_ws;
  _Float16* P  = (_Float16*)ws;                       // [2048,256] f16, 1 MiB
  float* s_fwd = (float*)(ws + (1u << 20));           // [8192]
  float* s_bwd = s_fwd + BB * NN * HH;                // [8192]
  float* v_soc = (float*)(ws + (1u << 20) + 65536);   // [2048,256] fp32, 2 MiB
  _Float16* Wh = (_Float16*)(ws + (1u << 20) + 65536 + (2u << 20));
  _Float16* We1h = Wh;            // 8192
  _Float16* We2h = Wh + 8192;     // 4096
  _Float16* Wv1h = Wh + 12288;    // 8192
  _Float16* Wv2h = Wh + 20480;    // 16384
  _Float16* Wv3h = Wh + 36864;    // 8192

  k_prep<<<176, 256, 0, stream>>>(We1, We2, Wv1, Wv2, Wv3, Wh);
  k_proj<<<128, 256, 0, stream>>>(v_self, W_proj, P);
  k_attn<<<32, 256, 0, stream>>>(P, a_fwd, a_bwd, s_fwd, s_bwd);
  k_edge<<<BB * NN * HH, 256, 0, stream>>>(P, s_fwd, s_bwd, We1h, be1, We2h,
                                           be2, v_soc, alpha_out);
  k_node<<<128, 256, 0, stream>>>(v_soc, Wv1h, bv1, Wv2h, bv2, Wv3h, bv3,
                                  v_out);
}
